// SGNHeadLSS_46849503265334
// MI455X (gfx1250) — matrix operand order repaired and checked by reading, not compiled, not run on hardware
//
#include <hip/hip_runtime.h>
#include <stdint.h>

#define NVOX 262144
#define CDIM 128

typedef __attribute__((ext_vector_type(16))) _Float16 v16h;
typedef __attribute__((ext_vector_type(8)))  float    v8f;
typedef __attribute__((ext_vector_type(4)))  float    f32x4;
typedef __attribute__((ext_vector_type(4)))  unsigned int u32x4;

union Frag16 {            // one WMMA 16-bit operand fragment (8 VGPRs / lane)
  v16h v;
  unsigned int u[8];
  u32x4 q[2];
};

// ---- f16 helpers: v_cvt_pk_rtz_f16_f32 packs two converts into one VALU op ----
static __device__ __forceinline__ unsigned int pack2h(float lo, float hi) {
  auto p = __builtin_amdgcn_cvt_pkrtz(lo, hi);      // 2 x _Float16 in one dword
  unsigned int u; __builtin_memcpy(&u, &p, sizeof(u)); return u;
}
static __device__ __forceinline__ unsigned short f2h(float f) {
  _Float16 h = (_Float16)f;                          // single v_cvt_f16_f32
  unsigned short u; __builtin_memcpy(&u, &h, 2); return u;
}

static __device__ __forceinline__ v8f wmma_f16(v16h a, v16h b, v8f c) {
  return __builtin_amdgcn_wmma_f32_16x16x32_f16(false, a, false, b, (short)0, c,
                                                false, false);
}

// A-matrix (16-bit 16x32) element i -> K offset (lanes 0-15 half=0, 16-31 half=1)
static __device__ __forceinline__ int a_kof(int i, int half) {
  return (i < 8) ? (8 * half + i) : (8 + 8 * half + i);
}

// wave-private LDS producer->consumer fence (DS ops of a wave are in-order)
static __device__ __forceinline__ void wave_lds_fence() {
  asm volatile("s_wait_dscnt 0" ::: "memory");
}

// ---- stage W[K][ncols] (f32 row-major) into fragment-native f16 LDS layout:
//      dst[((kt*NT+nt)*32 + lane)*16 + i],  elem i: K=kt*32+a_kof(i,lane>>4), n=nt*16+(lane&15)
static __device__ __forceinline__ void stage_bfrag(unsigned short* dst,
                                                   const float* __restrict__ w,
                                                   int KT, int NT, int ncols, int nvalid,
                                                   int tid, int nthr) {
  const int total = KT * NT * 512;
  for (int e = tid; e < total; e += nthr) {
    const int i    = e & 15;
    const int lane = (e >> 4) & 31;
    const int f    = e >> 9;
    const int nt   = f % NT;
    const int kt   = f / NT;
    const int k    = kt * 32 + a_kof(i, lane >> 4);
    const int n    = nt * 16 + (lane & 15);
    dst[e] = (n < nvalid) ? f2h(w[k * ncols + n]) : (unsigned short)0;
  }
}

// B fragment: two ds_load_b128 from fragment-native LDS layout
static __device__ __forceinline__ v16h load_bfrag(const unsigned short* base,
                                                  int kt, int NT, int nt, int lane) {
  Frag16 fr;
  const u32x4* p = (const u32x4*)(base + ((kt * NT + nt) * 32 + lane) * 16);
  fr.q[0] = p[0];
  fr.q[1] = p[1];
  return fr.v;
}

// A fragment from an f16 row in LDS (row-major, 16B-aligned, stride = multiple of 8)
static __device__ __forceinline__ v16h load_afrag_h16row(const unsigned short* row,
                                                         int kt, int half) {
  Frag16 fr;
  fr.q[0] = *(const u32x4*)(row + kt * 32 + 8 * half);
  fr.q[1] = *(const u32x4*)(row + kt * 32 + 16 + 8 * half);
  return fr.v;
}

// A fragment gathered from channel-major x3d (x3d[k*NVOX + n]): scalar loads, packed cvt
static __device__ __forceinline__ v16h load_afrag_gather(const float* __restrict__ x3d,
                                                         int kt, int half, int n) {
  Frag16 fr;
#pragma unroll
  for (int j = 0; j < 8; ++j) {
    const int k0 = kt * 32 + a_kof(2 * j, half);    // pairs never straddle the i=7/8 split
    fr.u[j] = pack2h(x3d[(size_t)k0 * NVOX + n], x3d[(size_t)(k0 + 1) * NVOX + n]);
  }
  return fr.v;
}

// ================= Kernel 0: per-voxel mask (1 = masked/prior, 0 = seed) =================
__global__ __launch_bounds__(256)
void mask_kernel(const int* __restrict__ midx, int nm,
                 const int* __restrict__ uidx, int nu,
                 unsigned char* __restrict__ mask) {
  const int i = blockIdx.x * 256 + threadIdx.x;
  if (i < nm)            mask[midx[i]] = 1;
  else if (i < nm + nu)  mask[uidx[i - nm]] = 0;
}

// ================= Kernel 1: aux head on seed rows (sem_logit) =================
__global__ __launch_bounds__(128)
void seed_aux_kernel(const float* __restrict__ x3d, const int* __restrict__ uidx,
                     const float* __restrict__ aux_w, const float* __restrict__ aux_b,
                     float* __restrict__ sem_out) {
  __shared__ unsigned short auxf[4 * 2 * 512];   // 8KB fragment-native f16
  const int tid  = threadIdx.x;
  const int wave = tid >> 5, lane = tid & 31;
  const int half = lane >> 4, ln16 = lane & 15;
  const int rowbase = (blockIdx.x * 4 + wave) * 16;

  stage_bfrag(auxf, aux_w, 4, 2, 20, 20, tid, 128);
  __syncthreads();

  const int myidx = uidx[rowbase + ln16];
  v16h a[4];
#pragma unroll
  for (int kt = 0; kt < 4; ++kt)
    a[kt] = load_afrag_gather(x3d, kt, half, myidx);

#pragma unroll
  for (int nt = 0; nt < 2; ++nt) {
    const int col = nt * 16 + ln16;
    const bool valid = (col < 20);
    v8f acc = {0, 0, 0, 0, 0, 0, 0, 0};
#pragma unroll
    for (int kt = 0; kt < 4; ++kt)
      acc = wmma_f16(a[kt], load_bfrag(auxf, kt, 2, nt, lane), acc);
    if (valid) {
      const float bias = aux_b[col];
#pragma unroll
      for (int r = 0; r < 8; ++r)
        sem_out[(size_t)(rowbase + r + 8 * half) * 20 + col] = acc[r] + bias;
    }
  }
}

// ================= Kernel 2: fully fused dense pipeline =================
// per 16-voxel tile: GEMM1 -> LN+leaky -> GEMM2 (prior) -> row select (mask)
//                    -> GEMM3 (sdb+leaky) -> GEMM4 (ssc) -> channel-major store
// Dynamic LDS layout (bytes):
//   [     0,16384) w1f   [16384,32768) w2f   [32768,49152) sdbf
//   [49152,53248) sscf   [53248,53760) ln g/b
//   [53760 + wave*6656): hbuf 4352B (f32 16x68; aliased LN-f16 / A3 rows, stride 136 us)
//                        dxbuf 2304B (f16 16x72 d-tile; aliased 20x20 f32 ssc staging)
#define FUSED_SMEM (53760 + 4 * 6656)

__global__ __launch_bounds__(128)
void fused_kernel(const float* __restrict__ x3d,
                  const float* __restrict__ w1, const float* __restrict__ b1,
                  const float* __restrict__ ln_g, const float* __restrict__ ln_b,
                  const float* __restrict__ w2, const float* __restrict__ b2,
                  const float* __restrict__ sdb_w, const float* __restrict__ sdb_b,
                  const float* __restrict__ ssc_w, const float* __restrict__ ssc_b,
                  const unsigned char* __restrict__ mask,
                  float* __restrict__ ssc_out) {
  extern __shared__ char smem[];
  unsigned short* w1f  = (unsigned short*)smem;
  unsigned short* w2f  = (unsigned short*)(smem + 16384);
  unsigned short* sdbf = (unsigned short*)(smem + 32768);
  unsigned short* sscf = (unsigned short*)(smem + 49152);
  float*          lng  = (float*)(smem + 53248);
  float*          lnb  = lng + 64;

  const int tid = threadIdx.x;
  stage_bfrag(w1f,  w1,    4, 4, 64,   64,   tid, 128);
  stage_bfrag(w2f,  w2,    2, 8, CDIM, CDIM, tid, 128);
  stage_bfrag(sdbf, sdb_w, 4, 4, 64,   64,   tid, 128);
  stage_bfrag(sscf, ssc_w, 2, 2, 20,   20,   tid, 128);
  for (int i = tid; i < 64; i += 128) { lng[i] = ln_g[i]; lnb[i] = ln_b[i]; }
  __syncthreads();

  const int wave = tid >> 5, lane = tid & 31;
  const int half = lane >> 4, ln16 = lane & 15;
  const int rowbase = (blockIdx.x * 4 + wave) * 16;

  char* wbase = smem + 53760 + wave * 6656;
  float*          hraw = (float*)wbase;            // 16 x 68 f32
  unsigned short* hus  = (unsigned short*)wbase;   // aliased f16 rows, stride 136
  unsigned short* dbuf = (unsigned short*)(wbase + 4352); // 16 x 72 f16
  float*          cbuf = (float*)(wbase + 4352);   // aliased 20 x 20 f32

  // ---- A1: dense channel-major gather (consecutive n across lanes -> coalesced)
  v16h a1[4];
#pragma unroll
  for (int kt = 0; kt < 4; ++kt)
    a1[kt] = load_afrag_gather(x3d, kt, half, rowbase + ln16);

  // ---- GEMM1: [16,128]@[128,64] (+b1) -> hraw (f32)
#pragma unroll
  for (int nt = 0; nt < 4; ++nt) {
    v8f acc = {0, 0, 0, 0, 0, 0, 0, 0};
#pragma unroll
    for (int kt = 0; kt < 4; ++kt)
      acc = wmma_f16(a1[kt], load_bfrag(w1f, kt, 4, nt, lane), acc);
    const float bias = b1[nt * 16 + ln16];
#pragma unroll
    for (int r = 0; r < 8; ++r)
      hraw[(r + 8 * half) * 68 + nt * 16 + ln16] = acc[r] + bias;
  }
  wave_lds_fence();

  // ---- LayerNorm(64)+leaky; lane<16 owns row `lane`, overwrites it with f16
  if (lane < 16) {
    float v[64];
    const float* row = &hraw[lane * 68];
#pragma unroll
    for (int j = 0; j < 16; ++j) {
      const f32x4 t = *(const f32x4*)(row + 4 * j);
      v[4 * j] = t.x; v[4 * j + 1] = t.y; v[4 * j + 2] = t.z; v[4 * j + 3] = t.w;
    }
    float mu = 0.f;
#pragma unroll
    for (int j = 0; j < 64; ++j) mu += v[j];
    mu *= (1.f / 64.f);
    float var = 0.f;
#pragma unroll
    for (int j = 0; j < 64; ++j) { const float d = v[j] - mu; var += d * d; }
    const float rs = rsqrtf(var * (1.f / 64.f) + 1e-5f);
    unsigned short* orow = hus + lane * 136;
#pragma unroll
    for (int j = 0; j < 8; ++j) {
      float t[8];
#pragma unroll
      for (int q = 0; q < 8; ++q) {
        float y = (v[8 * j + q] - mu) * rs * lng[8 * j + q] + lnb[8 * j + q];
        t[q] = (y > 0.f) ? y : 0.01f * y;
      }
      u32x4 qv;
      qv.x = pack2h(t[0], t[1]); qv.y = pack2h(t[2], t[3]);
      qv.z = pack2h(t[4], t[5]); qv.w = pack2h(t[6], t[7]);
      *(u32x4*)(orow + 8 * j) = qv;
    }
  }
  wave_lds_fence();

  // ---- A2 from LN rows, then (fence) the region is reused for the A3 rows
  v16h a2[2];
#pragma unroll
  for (int kt = 0; kt < 2; ++kt)
    a2[kt] = load_afrag_h16row(hus + ln16 * 136, kt, half);
  wave_lds_fence();

  // ---- GEMM2: [16,64]@[64,128] (+b2) -> prior rows as f16 (A3 layout, stride 136)
#pragma unroll
  for (int nt = 0; nt < 8; ++nt) {
    v8f acc = {0, 0, 0, 0, 0, 0, 0, 0};
#pragma unroll
    for (int kt = 0; kt < 2; ++kt)
      acc = wmma_f16(a2[kt], load_bfrag(w2f, kt, 8, nt, lane), acc);
    const float bias = b2[nt * 16 + ln16];
#pragma unroll
    for (int r = 0; r < 8; ++r)
      hus[(r + 8 * half) * 136 + nt * 16 + ln16] = f2h(acc[r] + bias);
  }
  wave_lds_fence();

  // ---- seed rows: overwrite prior with the original x3d row (L0/L2-hot re-read)
  if (lane < 16 && mask[rowbase + lane] == 0) {
    const int n = rowbase + lane;
    unsigned short* arow = hus + lane * 136;
    for (int j = 0; j < 16; ++j) {           // 8 channels per iteration (kept rolled)
      const int c = 8 * j;
      u32x4 qv;
      qv.x = pack2h(x3d[(size_t)(c + 0) * NVOX + n], x3d[(size_t)(c + 1) * NVOX + n]);
      qv.y = pack2h(x3d[(size_t)(c + 2) * NVOX + n], x3d[(size_t)(c + 3) * NVOX + n]);
      qv.z = pack2h(x3d[(size_t)(c + 4) * NVOX + n], x3d[(size_t)(c + 5) * NVOX + n]);
      qv.w = pack2h(x3d[(size_t)(c + 6) * NVOX + n], x3d[(size_t)(c + 7) * NVOX + n]);
      *(u32x4*)(arow + 8 * j) = qv;
    }
  }
  wave_lds_fence();

  // ---- A3 from the selected rows
  v16h a3[4];
#pragma unroll
  for (int kt = 0; kt < 4; ++kt)
    a3[kt] = load_afrag_h16row(hus + ln16 * 136, kt, half);

  // ---- GEMM3: d = leaky(row @ sdb_w + sdb_b) -> dbuf (f16)
#pragma unroll
  for (int nt = 0; nt < 4; ++nt) {
    v8f acc = {0, 0, 0, 0, 0, 0, 0, 0};
#pragma unroll
    for (int kt = 0; kt < 4; ++kt)
      acc = wmma_f16(a3[kt], load_bfrag(sdbf, kt, 4, nt, lane), acc);
    const float bias = sdb_b[nt * 16 + ln16];
#pragma unroll
    for (int r = 0; r < 8; ++r) {
      float y = acc[r] + bias;
      y = (y > 0.f) ? y : 0.01f * y;
      dbuf[(r + 8 * half) * 72 + nt * 16 + ln16] = f2h(y);
    }
  }
  wave_lds_fence();

  v16h a4[2];
#pragma unroll
  for (int kt = 0; kt < 2; ++kt)
    a4[kt] = load_afrag_h16row(dbuf + ln16 * 72, kt, half);
  wave_lds_fence();   // dbuf dead; region reused as cbuf

  // ---- GEMM4: ssc = d @ ssc_w + ssc_b -> cbuf (transposed staging)
#pragma unroll
  for (int nt = 0; nt < 2; ++nt) {
    const int col = nt * 16 + ln16;
    const bool valid = (col < 20);
    v8f acc = {0, 0, 0, 0, 0, 0, 0, 0};
#pragma unroll
    for (int kt = 0; kt < 2; ++kt)
      acc = wmma_f16(a4[kt], load_bfrag(sscf, kt, 2, nt, lane), acc);
    if (valid) {
      const float bias = ssc_b[col];
#pragma unroll
      for (int r = 0; r < 8; ++r)
        cbuf[col * 20 + r + 8 * half] = acc[r] + bias;
    }
  }
  wave_lds_fence();

  // ---- channel-major output: lane<20 owns one channel, 16 voxels as 4x b128
  if (lane < 20) {
    float* op = ssc_out + (size_t)lane * NVOX + rowbase;
    const float* cp = &cbuf[lane * 20];
#pragma unroll
    for (int j = 0; j < 4; ++j)
      *(f32x4*)(op + 4 * j) = *(const f32x4*)(cp + 4 * j);
  }
}

extern "C" void kernel_launch(void* const* d_in, const int* in_sizes, int n_in,
                              void* d_out, int out_size, void* d_ws, size_t ws_size,
                              hipStream_t stream) {
  (void)n_in; (void)out_size; (void)ws_size;
  const float* x3d   = (const float*)d_in[0];
  const float* w1    = (const float*)d_in[1];
  const float* b1    = (const float*)d_in[2];
  const float* ln_g  = (const float*)d_in[3];
  const float* ln_b  = (const float*)d_in[4];
  const float* w2    = (const float*)d_in[5];
  const float* b2    = (const float*)d_in[6];
  const float* sdb_w = (const float*)d_in[7];
  const float* sdb_b = (const float*)d_in[8];
  const float* ssc_w = (const float*)d_in[9];
  const float* ssc_b = (const float*)d_in[10];
  const float* aux_w = (const float*)d_in[11];
  const float* aux_b = (const float*)d_in[12];
  const int*   uidx  = (const int*)d_in[13];
  const int*   midx  = (const int*)d_in[14];

  float* out     = (float*)d_out;
  float* ssc_out = out;                      // [20, NVOX]
  float* sem_out = out + (size_t)20 * NVOX;  // [Nu, 20]
  unsigned char* mask = (unsigned char*)d_ws; // [NVOX] bytes

  const int nu = in_sizes[13];  // 65536
  const int nm = in_sizes[14];  // 196608

  mask_kernel    <<<(nm + nu) / 256, 256, 0, stream>>>(midx, nm, uidx, nu, mask);
  seed_aux_kernel<<<nu / 64, 128, 0, stream>>>(x3d, uidx, aux_w, aux_b, sem_out);
  fused_kernel   <<<NVOX / 64, 128, FUSED_SMEM, stream>>>(
      x3d, w1, b1, ln_g, ln_b, w2, b2, sdb_w, sdb_b, ssc_w, ssc_b, mask, ssc_out);
}
